// FluxModel_51256139711090
// MI455X (gfx1250) — matrix operand order repaired and checked by reading, not compiled, not run on hardware
//
#include <hip/hip_runtime.h>
#include <hip/hip_bf16.h>

// CDNA5 (gfx1250): analytic grad+Hessian propagation through a sin-MLP using
// v_wmma_f32_16x16x32_bf16. 16 points/workgroup, 10 derivative states/point ->
// per layer a (160x256)@(256x256) GEMM via WMMA. Loop nest: K-step outermost in
// the GEMM so each wave loads its 10 A-fragments (ds_load_b128) ONCE and reuses
// them for both of its N-tiles (20 persistent f32 accumulators). The sin/cos
// chain rule is lane-local across the accumulators.

typedef __attribute__((ext_vector_type(16))) __bf16 v16bf;
typedef __attribute__((ext_vector_type(8)))  __bf16 v8bf;
typedef __attribute__((ext_vector_type(8)))  float  v8f;

#define DIM      256
#define NLAYER   8
#define TPB      256      // threads per block (8 waves, wave32)
#define TPTS     16       // points per workgroup
#define NSTATE   10       // h, J0..J2, H00,H01,H02,H11,H12,H22
#define SPITCH   264      // padded row pitch (bf16 elems): 528B -> bank-friendly
#define SBUF_ELEMS (NSTATE * TPTS * SPITCH)     // 42240 bf16 per buffer
#define SMEM_BYTES (2 * SBUF_ELEMS * 2)         // 168960 B (double buffered)

__device__ inline v8f v8f_zero() {
  v8f r;
#pragma unroll
  for (int i = 0; i < 8; ++i) r[i] = 0.0f;
  return r;
}

// Transpose + convert hidden weights: wt[l][n][k] = (bf16) Ws[l][k][n]
__global__ void flux_prep_kernel(const float* __restrict__ Ws, __bf16* __restrict__ wt) {
  int idx = blockIdx.x * blockDim.x + threadIdx.x;   // 0 .. 8*256*256-1
  int l   = idx >> 16;
  int rem = idx & 0xFFFF;
  int n   = rem >> 8;
  int k   = rem & 0xFF;
  wt[((size_t)l << 16) + (n << 8) + k] = (__bf16)Ws[((size_t)l << 16) + (k << 8) + n];
}

__global__ void __launch_bounds__(TPB, 1)
flux_main_kernel(const float* __restrict__ x,
                 const float* __restrict__ W_in,
                 const float* __restrict__ b_in,
                 const float* __restrict__ bs,
                 const float* __restrict__ W_flux,
                 const float* __restrict__ W_bh,
                 const float* __restrict__ b_bh,
                 const __bf16* __restrict__ wts,
                 float* __restrict__ out,
                 int n_points) {
  extern __shared__ __align__(16) __bf16 S[];     // 2 state buffers, bf16
  __shared__ float red[TPTS * NSTATE];

  const int tid    = threadIdx.x;
  const int wave   = tid >> 5;
  const int lane   = tid & 31;
  const int block0 = blockIdx.x * TPTS;

  // ---------------- phase 0: input layer + derivative seeding ----------------
  // z = x @ W_in + b_in ; h=sin z ; J_i = cos(z)*Win_i ; H_ij = -sin(z)*Win_i*Win_j
  for (int idx = tid; idx < TPTS * DIM; idx += TPB) {
    int p = idx >> 8;
    int n = idx & 255;
    int gp = block0 + p;
    float x0 = x[gp * 3 + 0], x1 = x[gp * 3 + 1], x2 = x[gp * 3 + 2];
    float w0 = W_in[n], w1 = W_in[DIM + n], w2 = W_in[2 * DIM + n];
    float z  = fmaf(x0, w0, fmaf(x1, w1, fmaf(x2, w2, b_in[n])));
    float sh = __sinf(z), ch = __cosf(z);
    __bf16* S0 = S;   // buffer 0
    S0[(0 * TPTS + p) * SPITCH + n] = (__bf16)sh;
    S0[(1 * TPTS + p) * SPITCH + n] = (__bf16)(ch * w0);
    S0[(2 * TPTS + p) * SPITCH + n] = (__bf16)(ch * w1);
    S0[(3 * TPTS + p) * SPITCH + n] = (__bf16)(ch * w2);
    S0[(4 * TPTS + p) * SPITCH + n] = (__bf16)(-sh * w0 * w0);
    S0[(5 * TPTS + p) * SPITCH + n] = (__bf16)(-sh * w0 * w1);
    S0[(6 * TPTS + p) * SPITCH + n] = (__bf16)(-sh * w0 * w2);
    S0[(7 * TPTS + p) * SPITCH + n] = (__bf16)(-sh * w1 * w1);
    S0[(8 * TPTS + p) * SPITCH + n] = (__bf16)(-sh * w1 * w2);
    S0[(9 * TPTS + p) * SPITCH + n] = (__bf16)(-sh * w2 * w2);
  }
  __syncthreads();

  // ---------------- phase 1: 8 hidden layers via WMMA ----------------
  const int nlane = lane & 15;
  const int pbase = (lane >= 16) ? 8 : 0;    // C/D row offset
  const int aoff  = (lane >= 16) ? 8 : 0;    // A-frag K chunk offset
  const int boff  = (lane >= 16) ? 8 : 0;    // B-frag K offset

  for (int l = 0; l < NLAYER; ++l) {
    const __bf16* Scur = S + (size_t)(l & 1) * SBUF_ELEMS;
    __bf16*       Snxt = S + (size_t)((l + 1) & 1) * SBUF_ELEMS;
    const __bf16* wl   = wts + ((size_t)l << 16);
    const float*  bl   = bs + l * DIM;

    // prefetch next layer's weights into cache (global_prefetch_b8)
    if (l + 1 < NLAYER) {
      const char* wnext = (const char*)(wts + ((size_t)(l + 1) << 16));
#pragma unroll
      for (int c = 0; c < 8; ++c)
        __builtin_prefetch(wnext + (size_t)(tid + c * TPB) * 64, 0, 1);
    }

    // 20 persistent accumulators: [nt2][state]
    v8f acc[2][NSTATE];
#pragma unroll
    for (int nt2 = 0; nt2 < 2; ++nt2)
#pragma unroll
      for (int s2 = 0; s2 < NSTATE; ++s2) acc[nt2][s2] = v8f_zero();

#pragma unroll
    for (int ks = 0; ks < 8; ++ks) {          // K = 256 in steps of 32
      const int kbase = ks * 32;

      // Load the 10 A-fragments once (shared by both N-tiles): 2 ds_load_b128 each
      union AU { v16bf v; v8bf h[2]; } au[NSTATE];
#pragma unroll
      for (int s2 = 0; s2 < NSTATE; ++s2) {
        const __bf16* ap = Scur + (size_t)(s2 * TPTS + nlane) * SPITCH + kbase + aoff;
        au[s2].h[0] = *(const v8bf*)(ap);
        au[s2].h[1] = *(const v8bf*)(ap + 16);
      }

#pragma unroll
      for (int nt2 = 0; nt2 < 2; ++nt2) {
        const int ncol = (wave * 2 + nt2) * 16 + nlane;   // this lane's output dim
        // B fragment: Wt[n][k] contiguous in k -> two global 16B loads
        union { v16bf v; v8bf h[2]; } bu;
        const __bf16* wp = wl + ((size_t)ncol << 8) + kbase + boff;
        bu.h[0] = *(const v8bf*)(wp);
        bu.h[1] = *(const v8bf*)(wp + 16);
#pragma unroll
        for (int s2 = 0; s2 < NSTATE; ++s2) {
          acc[nt2][s2] = __builtin_amdgcn_wmma_f32_16x16x32_bf16(
              false, au[s2].v, false, bu.v, (short)0, acc[nt2][s2], false, false);
        }
      }
    }

#pragma unroll
    for (int nt2 = 0; nt2 < 2; ++nt2) {
      const int ncol = (wave * 2 + nt2) * 16 + nlane;

      // bias only on the h state (derivatives of a constant vanish)
      {
        float bz = bl[ncol];
#pragma unroll
        for (int v = 0; v < 8; ++v) acc[nt2][0][v] += bz;
      }

      // chain rule, lane-local (acc element v = point pbase+v, dim ncol)
#pragma unroll
      for (int v = 0; v < 8; ++v) {
        float z  = acc[nt2][0][v];
        float sh = __sinf(z), ch = __cosf(z);
        float j0 = acc[nt2][1][v], j1 = acc[nt2][2][v], j2 = acc[nt2][3][v];
        acc[nt2][4][v] = ch * acc[nt2][4][v] - sh * j0 * j0;
        acc[nt2][5][v] = ch * acc[nt2][5][v] - sh * j0 * j1;
        acc[nt2][6][v] = ch * acc[nt2][6][v] - sh * j0 * j2;
        acc[nt2][7][v] = ch * acc[nt2][7][v] - sh * j1 * j1;
        acc[nt2][8][v] = ch * acc[nt2][8][v] - sh * j1 * j2;
        acc[nt2][9][v] = ch * acc[nt2][9][v] - sh * j2 * j2;
        acc[nt2][1][v] = ch * j0;
        acc[nt2][2][v] = ch * j1;
        acc[nt2][3][v] = ch * j2;
        acc[nt2][0][v] = sh;
      }

      // store new states (bf16) into the other buffer
#pragma unroll
      for (int s2 = 0; s2 < NSTATE; ++s2) {
#pragma unroll
        for (int v = 0; v < 8; ++v)
          Snxt[(size_t)(s2 * TPTS + pbase + v) * SPITCH + ncol] = (__bf16)acc[nt2][s2][v];
      }
    }
    __syncthreads();
  }

  // ---------------- phase 2: heads + spherical assembly ----------------
  const __bf16* Sf = S;   // after 8 buffer flips, final states are in buffer 0
  if (tid < TPTS * NSTATE) {
    int p = tid / NSTATE;
    int o = tid % NSTATE;
    float acc = 0.0f;
    if (o < 8) {
      // o: 0..2 -> g_i (states 1..3); 3..7 -> H00,H01,H11,H02,H12
      int st = (o == 0) ? 1 : (o == 1) ? 2 : (o == 2) ? 3 :
               (o == 3) ? 4 : (o == 4) ? 5 : (o == 5) ? 7 :
               (o == 6) ? 6 : 8;
      for (int k = 0; k < DIM; ++k)
        acc += (float)Sf[(st * TPTS + p) * SPITCH + k] * W_flux[k];
    } else {
      int c = o - 8;
      for (int k = 0; k < DIM; ++k)
        acc += (float)Sf[(0 * TPTS + p) * SPITCH + k] * W_bh[k * 2 + c];
      acc += b_bh[c];
    }
    red[p * NSTATE + o] = acc;
  }
  __syncthreads();

  if (tid < TPTS) {
    int p  = tid;
    int gp = block0 + p;
    float x0 = x[gp * 3 + 0], x1 = x[gp * 3 + 1], x2 = x[gp * 3 + 2];
    float r  = sqrtf(x0 * x0 + x1 * x1 + x2 * x2);
    float th = acosf(x2 / r);
    float ph = atan2f(x1, x0);
    float sth = __sinf(th), cth = __cosf(th);
    float sph = __sinf(ph), cph = __cosf(ph);
    const float* rp = red + p * NSTATE;
    float g0 = rp[0], g1 = rp[1], g2 = rp[2];
    float H00 = rp[3], H01 = rp[4], H11 = rp[5], H02 = rp[6], H12 = rp[7];
    float bh0 = rp[8], bh1 = rp[9];

    float dflux_dr = g0 * sth * cph + g1 * sth * sph + g2 * cth;
    float b_r = -H00 * cth * cph * sph + H01 * cth * cph * cph
                - H01 * cth * sph * sph + H11 * cth * sph * cph
                + H02 * sth * sph - H12 * sth * cph;
    float vr = b_r, vt = bh0, vp = bh1;
    float bx = vr * sth * cph + vt * cth * cph - vp * sph;
    float by = vr * sth * sph + vt * cth * sph + vp * cph;
    float bz = vr * cth - vt * sth;

    out[gp * 3 + 0] = bx;
    out[gp * 3 + 1] = by;
    out[gp * 3 + 2] = bz;
    out[(size_t)3 * n_points + gp] = dflux_dr;
  }
}

extern "C" void kernel_launch(void* const* d_in, const int* in_sizes, int n_in,
                              void* d_out, int out_size, void* d_ws, size_t ws_size,
                              hipStream_t stream) {
  const float* x      = (const float*)d_in[0];
  const float* W_in   = (const float*)d_in[1];
  const float* b_in   = (const float*)d_in[2];
  const float* Ws     = (const float*)d_in[3];
  const float* bs     = (const float*)d_in[4];
  const float* W_flux = (const float*)d_in[5];
  const float* b_flux = (const float*)d_in[6]; (void)b_flux;   // cancels in all outputs
  const float* W_bh   = (const float*)d_in[7];
  const float* b_bh   = (const float*)d_in[8];
  float* out = (float*)d_out;
  (void)n_in; (void)out_size; (void)ws_size;

  const int n_points = in_sizes[0] / 3;          // x is (N,3)
  __bf16* wt = (__bf16*)d_ws;                    // 8*256*256 bf16 = 1 MB

  // 1) transpose+convert hidden weights to bf16 (same stream -> ordered)
  flux_prep_kernel<<<(NLAYER * DIM * DIM) / TPB, TPB, 0, stream>>>(Ws, wt);

  // 2) main fused kernel; 165 KB dynamic LDS (CDNA5 WGP has 320 KB)
  hipFuncSetAttribute(reinterpret_cast<const void*>(flux_main_kernel),
                      hipFuncAttributeMaxDynamicSharedMemorySize, SMEM_BYTES);
  flux_main_kernel<<<n_points / TPTS, TPB, SMEM_BYTES, stream>>>(
      x, W_in, b_in, bs, W_flux, W_bh, b_bh, wt, out, n_points);
}
